// GPSLayer_38062000177346
// MI455X (gfx1250) — compile-verified, matrix operand order checked by reading, not run on hardware
//
#include <hip/hip_runtime.h>
#include <hip/hip_bf16.h>

// ---------------------------------------------------------------------------
// GPS layer (GINE + global attention + FFN, 3x BatchNorm) for gfx1250.
// Matrix math runs on v_wmma_f32_16x16x32_bf16 (f32 accumulate).
// N=32768 nodes, D=256, G=64 graphs x S=512 nodes, H=8 heads, DH=32, E=524288.
// GEMMs are N-register-blocked: one wave owns a 16x64 C strip (4 WMMA chain
// per K-step, A fragment loaded once -> 4x less activation re-fetch).
// ---------------------------------------------------------------------------

#define NN 32768
#define DD 256
#define GG 64
#define SS 512
#define HH 8
#define DH 32
#define EE 524288
#define EPS 1e-5f

typedef __attribute__((ext_vector_type(16))) __bf16 v16bf;
typedef __attribute__((ext_vector_type(8)))  float  v8f;

// ---------------------------------------------------------------------------
// Edge phase: msg = relu(x[src] + edge_attr); agg[dst] += msg   (bandwidth)
// ---------------------------------------------------------------------------
__global__ __launch_bounds__(256)
void edge_scatter(const float* __restrict__ x, const int* __restrict__ ei,
                  const float* __restrict__ ea, float* __restrict__ agg)
{
    int idx = blockIdx.x * 256 + threadIdx.x;      // E*64 float4 chunks
    if (idx >= EE * 64) return;
    int e = idx >> 6;
    int c = idx & 63;                              // float4 index within row
    int src = ei[e];
    int dst = ei[EE + e];
    float4 xv = ((const float4*)x)[(size_t)src * 64 + c];
    float4 av = ((const float4*)ea)[(size_t)e * 64 + c];
    float4 m;
    m.x = fmaxf(xv.x + av.x, 0.f);
    m.y = fmaxf(xv.y + av.y, 0.f);
    m.z = fmaxf(xv.z + av.z, 0.f);
    m.w = fmaxf(xv.w + av.w, 0.f);
    float* b = agg + (size_t)dst * DD + c * 4;
    unsafeAtomicAdd(b + 0, m.x);
    unsafeAtomicAdd(b + 1, m.y);
    unsafeAtomicAdd(b + 2, m.z);
    unsafeAtomicAdd(b + 3, m.w);
}

// ---------------------------------------------------------------------------
// bf16 WMMA GEMM: C[M,Nout] = epi(A[M,K] (+A2) @ B + bias (+Res))
// One wave computes a 16x64 strip of C (4 N-tiles); K-loop in steps of 32.
// Per K-step: 1 A-fragment load, then 4 back-to-back WMMAs vs 4 B fragments.
// BT=false: B is row-major [K,Nout] (gin_w*, ff_w*).
// BT=true : B is row-major [Nout,K], used transposed (in_w, out_w).
// Fragment layouts per ISA 7.12.2 (wave32, 16-bit A/B, f32 C/D).
// ---------------------------------------------------------------------------
#define NB 4
template<bool ADD2, bool BT, bool RELU, bool RES>
__global__ __launch_bounds__(32)
void gemm_bf16(const float* __restrict__ A, const float* __restrict__ A2,
               const float* __restrict__ B, const float* __restrict__ bias,
               const float* __restrict__ Res, float* __restrict__ C,
               int K, int Nout)
{
    const int lane = threadIdx.x;
    const int tn = blockIdx.x * 16 * NB;
    const int tm = blockIdx.y * 16;
    const int r  = lane & 15;                 // row of A frag / col of B frag
    const int kb = (lane >> 4) << 3;          // 0 or 8 : K sub-offset

    const float* Arow  = A  + (size_t)(tm + r) * K;
    const float* A2row = A2 + (size_t)(tm + r) * K;   // only read if ADD2

    v8f acc[NB];
#pragma unroll
    for (int j = 0; j < NB; ++j) acc[j] = (v8f){};

    for (int k0 = 0; k0 < K; k0 += 32) {
        v16bf a;
#pragma unroll
        for (int i = 0; i < 8; ++i) {
            float lo = Arow[k0 + kb + i];
            float hi = Arow[k0 + kb + 16 + i];
            if (ADD2) {
                lo += A2row[k0 + kb + i];
                hi += A2row[k0 + kb + 16 + i];
            }
            a[i]     = (__bf16)lo;
            a[i + 8] = (__bf16)hi;
        }
        if (k0 + 32 < K)
            __builtin_prefetch(&Arow[k0 + 32], 0, 3);
#pragma unroll
        for (int j = 0; j < NB; ++j) {
            v16bf b;
            if (BT) {
                const float* Bcol = B + (size_t)(tn + j * 16 + r) * K;
#pragma unroll
                for (int i = 0; i < 8; ++i) {
                    b[i]     = (__bf16)Bcol[k0 + kb + i];
                    b[i + 8] = (__bf16)Bcol[k0 + kb + 16 + i];
                }
            } else {
#pragma unroll
                for (int i = 0; i < 8; ++i) {
                    b[i]     = (__bf16)B[(size_t)(k0 + kb + i)      * Nout + tn + j * 16 + r];
                    b[i + 8] = (__bf16)B[(size_t)(k0 + kb + 16 + i) * Nout + tn + j * 16 + r];
                }
            }
            acc[j] = __builtin_amdgcn_wmma_f32_16x16x32_bf16(
                         false, a, false, b, (short)0, acc[j], false, false);
        }
    }
    // epilogue: D-tile element (m,n): lane = n + 16*(m>=8), vgpr = m&7
    const int mb = tm + kb;
#pragma unroll
    for (int j = 0; j < NB; ++j) {
        const int   n  = tn + j * 16 + r;
        const float bn = bias[n];
#pragma unroll
        for (int rr = 0; rr < 8; ++rr) {
            int   m = mb + rr;
            float v = acc[j][rr] + bn;
            if (RELU) v = fmaxf(v, 0.f);
            if (RES)  v += Res[(size_t)m * Nout + n];
            C[(size_t)m * Nout + n] = v;
        }
    }
}

// ---------------------------------------------------------------------------
// Global attention: one wave per (graph, head, 16-query tile).
// scores (16x512) via 32 WMMAs (K=DH=32 -> one WMMA per key tile) -> LDS,
// two-pass softmax, ctx = P @ V via 2x16 WMMAs reading P back as bf16.
// ---------------------------------------------------------------------------
__global__ __launch_bounds__(32)
void attn_kernel(const float* __restrict__ qkv, float* __restrict__ ctx)
{
    __shared__ float sc[16][SS];                 // 32 KB score/prob strip
    const int lane = threadIdx.x;
    const int qt = blockIdx.x;                   // query tile 0..31
    const int h  = blockIdx.y;                   // head 0..7
    const int g  = blockIdx.z;                   // graph 0..63
    const int r  = lane & 15;
    const int hi = lane >> 4;
    const int kb = hi << 3;
    const int row0 = g * SS;
    const size_t ld = 3 * DD;                    // qkv row stride = 768

    // Q fragment (row = qt*16+r, dims = this head's 32 columns), loaded once
    const float* qrow = qkv + (size_t)(row0 + qt * 16 + r) * ld + h * DH;
    v16bf aq;
#pragma unroll
    for (int i = 0; i < 8; ++i) {
        aq[i]     = (__bf16)qrow[kb + i];
        aq[i + 8] = (__bf16)qrow[kb + 16 + i];
    }
    const float scale = 0.17677669529663687f;    // 1/sqrt(32)
    for (int tt = 0; tt < SS / 16; ++tt) {
        const float* krow = qkv + (size_t)(row0 + tt * 16 + r) * ld + DD + h * DH;
        v16bf bk;
#pragma unroll
        for (int i = 0; i < 8; ++i) {
            bk[i]     = (__bf16)krow[kb + i];
            bk[i + 8] = (__bf16)krow[kb + 16 + i];
        }
        v8f sa = {};
        sa = __builtin_amdgcn_wmma_f32_16x16x32_bf16(
                 false, aq, false, bk, (short)0, sa, false, false);
#pragma unroll
        for (int rr = 0; rr < 8; ++rr)
            sc[rr + kb][tt * 16 + r] = sa[rr] * scale;
    }
    __syncthreads();

    // softmax over 512 keys: lane pair (r, r+16) splits the row
    {
        float m = -3.0e38f;
        const int c0 = hi * 256;
        for (int c = 0; c < 256; ++c) m = fmaxf(m, sc[r][c0 + c]);
        m = fmaxf(m, __shfl_xor(m, 16, 32));
        float s = 0.f;
        for (int c = 0; c < 256; ++c) {
            float e = __expf(sc[r][c0 + c] - m);
            sc[r][c0 + c] = e;
            s += e;
        }
        s += __shfl_xor(s, 16, 32);
        float inv = 1.f / s;
        for (int c = 0; c < 256; ++c) sc[r][c0 + c] *= inv;
    }
    __syncthreads();

    // ctx = P @ V : two 16-wide d tiles, 16 K-steps of 32
    for (int nt = 0; nt < 2; ++nt) {
        v8f acc = {};
        for (int ks = 0; ks < SS / 32; ++ks) {
            v16bf a, b;
#pragma unroll
            for (int i = 0; i < 8; ++i) {
                a[i]     = (__bf16)sc[r][ks * 32 + kb + i];
                a[i + 8] = (__bf16)sc[r][ks * 32 + kb + 16 + i];
                b[i]     = (__bf16)qkv[(size_t)(row0 + ks * 32 + kb + i) * ld
                                       + 2 * DD + h * DH + nt * 16 + r];
                b[i + 8] = (__bf16)qkv[(size_t)(row0 + ks * 32 + kb + 16 + i) * ld
                                       + 2 * DD + h * DH + nt * 16 + r];
            }
            acc = __builtin_amdgcn_wmma_f32_16x16x32_bf16(
                      false, a, false, b, (short)0, acc, false, false);
        }
#pragma unroll
        for (int rr = 0; rr < 8; ++rr) {
            int m = qt * 16 + rr + kb;
            ctx[(size_t)(row0 + m) * DD + h * DH + nt * 16 + r] = acc[rr];
        }
    }
}

// ---------------------------------------------------------------------------
// BatchNorm: per-channel sum & sumsq (atomic accumulate), then normalize.
// ---------------------------------------------------------------------------
__global__ __launch_bounds__(256)
void bn_stats(const float* __restrict__ X, float* __restrict__ stats)
{
    int c  = threadIdx.x;                        // channel
    int r0 = blockIdx.x * 128;                   // 128 rows per block
    float s = 0.f, ss = 0.f;
    for (int r = 0; r < 128; ++r) {
        float v = X[(size_t)(r0 + r) * DD + c];
        s += v;
        ss += v * v;
    }
    unsafeAtomicAdd(&stats[c], s);
    unsafeAtomicAdd(&stats[DD + c], ss);
}

__device__ __forceinline__ float bn_apply(float v, const float* st, int c,
                                          const float* g, const float* b)
{
    const float invN = 1.f / (float)NN;
    float mu  = st[c] * invN;
    float var = st[DD + c] * invN - mu * mu;
    return (v - mu) * rsqrtf(var + EPS) * g[c] + b[c];
}

__global__ __launch_bounds__(256)
void bn_combine(const float* __restrict__ s1, const float* __restrict__ s2,
                const float* __restrict__ st1, const float* __restrict__ st2,
                const float* __restrict__ g1, const float* __restrict__ b1,
                const float* __restrict__ g2, const float* __restrict__ b2,
                float* __restrict__ out)
{
    size_t idx = (size_t)blockIdx.x * 256 + threadIdx.x;   // N*D elements
    int c = (int)(idx & (DD - 1));
    out[idx] = bn_apply(s1[idx], st1, c, g1, b1)
             + bn_apply(s2[idx], st2, c, g2, b2);
}

__global__ __launch_bounds__(256)
void bn_final(const float* __restrict__ s3, const float* __restrict__ st,
              const float* __restrict__ g, const float* __restrict__ b,
              float* __restrict__ out)
{
    size_t idx = (size_t)blockIdx.x * 256 + threadIdx.x;
    int c = (int)(idx & (DD - 1));
    out[idx] = bn_apply(s3[idx], st, c, g, b);
}

// ---------------------------------------------------------------------------
// Host-side pipeline
// ---------------------------------------------------------------------------
extern "C" void kernel_launch(void* const* d_in, const int* in_sizes, int n_in,
                              void* d_out, int out_size, void* d_ws, size_t ws_size,
                              hipStream_t stream)
{
    const float* x       = (const float*)d_in[0];
    const int*   ei      = (const int*)  d_in[1];
    const float* ea      = (const float*)d_in[2];
    const float* gin_w1  = (const float*)d_in[3];
    const float* gin_b1  = (const float*)d_in[4];
    const float* gin_w2  = (const float*)d_in[5];
    const float* gin_b2  = (const float*)d_in[6];
    const float* in_w    = (const float*)d_in[7];
    const float* in_b    = (const float*)d_in[8];
    const float* out_w   = (const float*)d_in[9];
    const float* out_b   = (const float*)d_in[10];
    const float* ff_w1   = (const float*)d_in[11];
    const float* ff_b1   = (const float*)d_in[12];
    const float* ff_w2   = (const float*)d_in[13];
    const float* ff_b2   = (const float*)d_in[14];
    const float* bn1_g   = (const float*)d_in[15];
    const float* bn1_b   = (const float*)d_in[16];
    const float* bn2_g   = (const float*)d_in[17];
    const float* bn2_b   = (const float*)d_in[18];
    const float* bn3_g   = (const float*)d_in[19];
    const float* bn3_b   = (const float*)d_in[20];
    float* out = (float*)d_out;

    // workspace layout (floats)
    float* w     = (float*)d_ws;
    float* qkv   = w;                               // N*768
    float* bufA  = qkv  + (size_t)NN * 3 * DD;      // N*512 (t1, then ff1)
    float* agg   = bufA + (size_t)NN * 2 * DD;      // N*256 (agg, then s2)
    float* s1    = agg  + (size_t)NN * DD;          // N*256 (s1, then s3)
    float* ctxb  = s1   + (size_t)NN * DD;          // N*256 (ctx, then comb)
    float* stats = ctxb + (size_t)NN * DD;          // 3 * 512 floats
    float* st1 = stats, *st2 = stats + 2 * DD, *st3 = stats + 4 * DD;

    // zero accumulators (harness does not re-poison between replays)
    hipMemsetAsync(agg,   0, (size_t)NN * DD * sizeof(float), stream);
    hipMemsetAsync(stats, 0, 6 * DD * sizeof(float), stream);

    const dim3 wave(32);
    const dim3 gM16(DD / (16 * NB), NN / 16);       // Nout = 256 GEMMs
    const dim3 gQKV(3 * DD / (16 * NB), NN / 16);   // Nout = 768
    const dim3 gFF1(2 * DD / (16 * NB), NN / 16);   // Nout = 512

    // 1. GINE message scatter
    edge_scatter<<<EE * 64 / 256, 256, 0, stream>>>(x, ei, ea, agg);

    // 2. qkv = x @ in_w^T + in_b
    gemm_bf16<false, true, false, false><<<gQKV, wave, 0, stream>>>(
        x, nullptr, in_w, in_b, nullptr, qkv, DD, 3 * DD);

    // 3. attention -> ctx
    attn_kernel<<<dim3(SS / 16, HH, GG), wave, 0, stream>>>(qkv, ctxb);

    // 4. t1 = relu((x + agg) @ gin_w1 + gin_b1)
    gemm_bf16<true, false, true, false><<<gM16, wave, 0, stream>>>(
        x, agg, gin_w1, gin_b1, nullptr, bufA, DD, DD);

    // 5. s1 = x + t1 @ gin_w2 + gin_b2
    gemm_bf16<false, false, false, true><<<gM16, wave, 0, stream>>>(
        bufA, nullptr, gin_w2, gin_b2, x, s1, DD, DD);

    // 6. s2 = x + ctx @ out_w^T + out_b   (reuse agg buffer)
    gemm_bf16<false, true, false, true><<<gM16, wave, 0, stream>>>(
        ctxb, nullptr, out_w, out_b, x, agg, DD, DD);

    // 7. BN stats + combined normalize: comb = bn1(s1) + bn2(s2)
    bn_stats<<<NN / 128, 256, 0, stream>>>(s1, st1);
    bn_stats<<<NN / 128, 256, 0, stream>>>(agg, st2);
    bn_combine<<<(size_t)NN * DD / 256, 256, 0, stream>>>(
        s1, agg, st1, st2, bn1_g, bn1_b, bn2_g, bn2_b, ctxb);

    // 8. ff1 = relu(comb @ ff_w1 + ff_b1)
    gemm_bf16<false, false, true, false><<<gFF1, wave, 0, stream>>>(
        ctxb, nullptr, ff_w1, ff_b1, nullptr, bufA, DD, 2 * DD);

    // 9. s3 = comb + ff1 @ ff_w2 + ff_b2
    gemm_bf16<false, false, false, true><<<gM16, wave, 0, stream>>>(
        bufA, nullptr, ff_w2, ff_b2, ctxb, s1, 2 * DD, DD);

    // 10. out = bn3(s3)
    bn_stats<<<NN / 128, 256, 0, stream>>>(s1, st3);
    bn_final<<<(size_t)NN * DD / 256, 256, 0, stream>>>(
        s1, st3, bn3_g, bn3_b, out);
}
